// MoEINR_12429635354820
// MI455X (gfx1250) — matrix operand order repaired and checked by reading, not compiled, not run on hardware
//
#include <hip/hip_runtime.h>

typedef __attribute__((ext_vector_type(16))) _Float16 v16h;
typedef __attribute__((ext_vector_type(8)))  _Float16 v8h;
typedef __attribute__((ext_vector_type(8)))  float    v8f;

union AV { v16h v; v8h h[2]; };

// sin(30*x) == v_sin_f32(x * 30/(2*pi));  v_sin input is in revolutions.
#define OMEGA_REV 4.77464829275686f   // 30 / (2*pi)
#define INV_2PI   0.15915494309189535f

// ---------------- LDS layout (element units) ----------------
// Weight tiles stored [K][N] (B-fragment friendly), strides padded to
// multiples of 8 halfs (16B) for aligned ds_load_b128 + bank spread.
constexpr int S1_STR = 136;   // N=128
constexpr int S2_STR = 264;   // N=256
constexpr int F1_STR = 72;    // N=64
constexpr int F2_STR = 72;    // N=64
constexpr int F3_STR = 264;   // N=256
constexpr int P2_STR = 40;    // N=32  (policy layer 2)
constexpr int GE_STR = 24;    // N=16  (gate+experts combined)
constexpr int SSTR   = 136;   // per-wave scratch: [16][<=128]
constexpr int HSTR   = 264;   // per-wave h/enc_feat: [16][256]
constexpr int GSTR   = 20;    // per-wave f32 gate/expert logits [16][20]

constexpr int TILES_PER_WAVE = 4;     // amortize weight preload over 4 tiles
constexpr int ROWS_PER_BLOCK = 128 * TILES_PER_WAVE;  // 512

constexpr int OFF_WS1 = 0;                       // halfs
constexpr int OFF_WS2 = OFF_WS1 + 64  * S1_STR;  // 8704
constexpr int OFF_WF1 = OFF_WS2 + 128 * S2_STR;  // 42496
constexpr int OFF_WF2 = OFF_WF1 + 256 * F1_STR;  // 60928
constexpr int OFF_WF3 = OFF_WF2 + 64  * F2_STR;  // 65536
constexpr int OFF_WP2 = OFF_WF3 + 64  * F3_STR;  // 82432
constexpr int OFF_WGE = OFF_WP2 + 32  * P2_STR;  // 83712
constexpr int HALFS_TOTAL = OFF_WGE + 288 * GE_STR;   // 90624 halfs
constexpr int F32_BASE_BYTES = HALFS_TOTAL * 2;       // 181248

// float-typed constants region (offsets in floats)
constexpr int FO_BS1 = 0;      // s1_b   [128]
constexpr int FO_BS2 = 128;    // s2_b   [256]
constexpr int FO_BF1 = 384;    // fc1_b  [64]
constexpr int FO_BF2 = 448;    // fc2_b  [64]
constexpr int FO_BF3 = 512;    // fc3_b  [256]
constexpr int FO_PEW = 768;    // pe_w   [32*4]
constexpr int FO_PEB = 896;    // pe_b   [32]
constexpr int FO_P1W = 928;    // p1_w   [32*4]
constexpr int FO_P1B = 1056;   // p1_b   [32]
constexpr int FO_P2B = 1088;   // p2_b   [32]
constexpr int FO_GEB = 1120;   // combined gate_b/exp_b [16]
constexpr int F32_COUNT = 1136;

constexpr int STG_BASE_BYTES = F32_BASE_BYTES + F32_COUNT * 4;           // 185792
constexpr int STG_PER_WAVE_BYTES = (16 * SSTR + 16 * HSTR) * 2 + 16 * GSTR * 4; // 14080
constexpr size_t SMEM_BYTES = (size_t)STG_BASE_BYTES + 8 * STG_PER_WAVE_BYTES;  // 298432 (<320KB/WGP)

// ---------------- cooperative preload helpers ----------------
// global weight is [N][K] row-major; LDS wants [K][STRIDE] f16
template<int K, int N, int STR>
__device__ __forceinline__ void preW(const float* __restrict__ g,
                                     _Float16* __restrict__ d, int t) {
  for (int i = t; i < K * N; i += 256) {
    int k = i / N, n = i - k * N;
    d[k * STR + n] = (_Float16)g[n * K + k];
  }
}
__device__ __forceinline__ void preF(const float* __restrict__ g,
                                     float* __restrict__ d, int n, int t) {
  for (int i = t; i < n; i += 256) d[i] = g[i];
}

// ---------------- WMMA GEMM stage ----------------
// C[16,N] = act( A[16,K] @ B[K,N] + bias ), all operands staged in LDS f16.
// ACT: 0 = sin(30*v), 1 = relu(v), 2 = relu(v + residual) (residual/out alias ok)
template<int K, int N, int BS, int AS, int OS, int ACT>
__device__ __forceinline__ void gemm_stage(const _Float16* __restrict__ A,
                                           const _Float16* __restrict__ Bw,
                                           const float* __restrict__ bias,
                                           _Float16* __restrict__ Out,
                                           const _Float16* __restrict__ Res,
                                           int rs, int lane) {
  constexpr int KS = K / 32;
  constexpr int NT = N / 16;
  const int l16 = lane & 15;
  const int hs  = lane >> 4;

  // Hoist all A fragments (ISA 16x32 f16 A layout: lanes 0-15 hold K 0-7 &
  // 16-23 of row M=lane; lanes 16-31 hold K 8-15 & 24-31 of row M=lane-16).
  AV a[KS];
#pragma unroll
  for (int ks = 0; ks < KS; ++ks) {
    const _Float16* ap = A + l16 * AS + 32 * ks + 8 * hs;
    a[ks].h[0] = *(const v8h*)(ap);
    a[ks].h[1] = *(const v8h*)(ap + 16);
  }
#pragma unroll
  for (int nt = 0; nt < NT; ++nt) {
    const int n0 = nt * 16;
    v8f acc = {0.f, 0.f, 0.f, 0.f, 0.f, 0.f, 0.f, 0.f};
#pragma unroll
    for (int ks = 0; ks < KS; ++ks) {
      // B fragment: lane = N col (n0 + l16); K rows 8*hs+s and 16+8*hs+s.
      const _Float16* bp = Bw + (32 * ks + 8 * hs) * BS + n0 + l16;
      AV b;
#pragma unroll
      for (int s = 0; s < 8; ++s) { b.v[s] = bp[s * BS]; b.v[8 + s] = bp[(16 + s) * BS]; }
      acc = __builtin_amdgcn_wmma_f32_16x16x32_f16(false, a[ks].v, false, b.v,
                                                   (short)0, acc, false, false);
    }
    // C layout: lanes 0-15 -> M = r, N = n0+lane; lanes 16-31 -> M = 8+r.
    const float bv = bias[n0 + l16];
    const int col = n0 + l16;
#pragma unroll
    for (int r = 0; r < 8; ++r) {
      const int row = r + 8 * hs;
      float v = acc[r] + bv;
      if constexpr (ACT == 0)      v = __builtin_amdgcn_sinf(v * OMEGA_REV);
      else if constexpr (ACT == 1) v = fmaxf(v, 0.0f);
      else                         v = fmaxf(v + (float)Res[row * rs + col], 0.0f);
      Out[row * OS + col] = (_Float16)v;
    }
  }
}

// ---------------- fused MoE-INR forward ----------------
__global__ __launch_bounds__(256) void moe_inr_kernel(
    const float* __restrict__ x,
    const float* __restrict__ pe_w,  const float* __restrict__ pe_b,
    const float* __restrict__ s1_w,  const float* __restrict__ s1_b,
    const float* __restrict__ s2_w,  const float* __restrict__ s2_b,
    const float* __restrict__ fc1_w, const float* __restrict__ fc1_b,
    const float* __restrict__ fc2_w, const float* __restrict__ fc2_b,
    const float* __restrict__ fc3_w, const float* __restrict__ fc3_b,
    const float* __restrict__ p1_w,  const float* __restrict__ p1_b,
    const float* __restrict__ p2_w,  const float* __restrict__ p2_b,
    const float* __restrict__ gate_w, const float* __restrict__ gate_b,
    const float* __restrict__ exp_w,  const float* __restrict__ exp_b,
    float* __restrict__ out, int Btot) {
  extern __shared__ __align__(16) char smem[];
  _Float16* Wh = (_Float16*)smem;
  float*    Wf = (float*)(smem + F32_BASE_BYTES);

  const int t    = threadIdx.x;
  const int lane = t & 31;
  const int wave = t >> 5;

  // ---- cooperative weight stage (f32 global -> f16 LDS, transposed) ----
  preW<64, 128, S1_STR>(s1_w,  Wh + OFF_WS1, t);
  preW<128, 256, S2_STR>(s2_w, Wh + OFF_WS2, t);
  preW<256, 64, F1_STR>(fc1_w, Wh + OFF_WF1, t);
  preW<64, 64, F2_STR>(fc2_w,  Wh + OFF_WF2, t);
  preW<64, 256, F3_STR>(fc3_w, Wh + OFF_WF3, t);
  preW<32, 32, P2_STR>(p2_w,   Wh + OFF_WP2, t);
  // combined gate+expert B matrix: K=288 ([enc|f2]), N=16
  // cols 0..6 = gate logits, 7..13 = expert preds (f2 rows zero), 14..15 = 0
  for (int i = t; i < 288 * 16; i += 256) {
    const int k = i >> 4, n = i & 15;
    float v = 0.f;
    if (n < 7)                  v = gate_w[n * 288 + k];
    else if (n < 14 && k < 256) v = exp_w[(n - 7) * 256 + k];
    Wh[OFF_WGE + k * GE_STR + n] = (_Float16)v;
  }
  preF(s1_b,  Wf + FO_BS1, 128, t);
  preF(s2_b,  Wf + FO_BS2, 256, t);
  preF(fc1_b, Wf + FO_BF1, 64,  t);
  preF(fc2_b, Wf + FO_BF2, 64,  t);
  preF(fc3_b, Wf + FO_BF3, 256, t);
  preF(pe_w,  Wf + FO_PEW, 128, t);
  preF(pe_b,  Wf + FO_PEB, 32,  t);
  preF(p1_w,  Wf + FO_P1W, 128, t);
  preF(p1_b,  Wf + FO_P1B, 32,  t);
  preF(p2_b,  Wf + FO_P2B, 32,  t);
  if (t < 16) {
    float v = 0.f;
    if (t < 7)       v = gate_b[t];
    else if (t < 14) v = exp_b[t - 7];
    Wf[FO_GEB + t] = v;
  }
  __syncthreads();   // only barrier: waves are independent from here on

  _Float16* S  = (_Float16*)(smem + STG_BASE_BYTES + wave * STG_PER_WAVE_BYTES);
  _Float16* H  = S + 16 * SSTR;
  float*    GE = (float*)(H + 16 * HSTR);

  const int l16 = lane & 15;
  const int hs  = lane >> 4;

  // ---- 4 independent 16-row tiles per wave (amortizes the preload) ----
#pragma clang loop unroll(disable)
  for (int tile = 0; tile < TILES_PER_WAVE; ++tile) {
    const int rowG = blockIdx.x * ROWS_PER_BLOCK + tile * 128 + wave * 16 + l16;

    // ---- stage 0: Fourier PE  h0 = [sin(u), cos(u)], u = x@pe_w.T + pe_b --
    float4 xv = make_float4(0.f, 0.f, 0.f, 0.f);
    if (rowG < Btot) xv = *(const float4*)(x + (size_t)rowG * 4);
#pragma unroll
    for (int j = 0; j < 16; ++j) {
      const int c = 16 * hs + j;
      const float u = xv.x * Wf[FO_PEW + c * 4 + 0] + xv.y * Wf[FO_PEW + c * 4 + 1] +
                      xv.z * Wf[FO_PEW + c * 4 + 2] + xv.w * Wf[FO_PEW + c * 4 + 3] +
                      Wf[FO_PEB + c];
      const float ur = u * INV_2PI;
      S[l16 * SSTR + c]      = (_Float16)__builtin_amdgcn_sinf(ur);
      S[l16 * SSTR + c + 32] = (_Float16)__builtin_amdgcn_cosf(ur);
    }

    // ---- SIREN encoder + bottleneck, all on WMMA ----
    gemm_stage<64, 128, S1_STR, SSTR, SSTR, 0>(S, Wh + OFF_WS1, Wf + FO_BS1, S, nullptr, 0, lane);
    gemm_stage<128, 256, S2_STR, SSTR, HSTR, 0>(S, Wh + OFF_WS2, Wf + FO_BS2, H, nullptr, 0, lane);
    gemm_stage<256, 64, F1_STR, HSTR, SSTR, 1>(H, Wh + OFF_WF1, Wf + FO_BF1, S, nullptr, 0, lane);
    gemm_stage<64, 64, F2_STR, SSTR, SSTR, 1>(S, Wh + OFF_WF2, Wf + FO_BF2, S, nullptr, 0, lane);
    // fc3 + skip + relu -> enc_feat, written over H in place
    gemm_stage<64, 256, F3_STR, SSTR, HSTR, 2>(S, Wh + OFF_WF3, Wf + FO_BF3, H, H, HSTR, lane);

    // ---- policy layer 1 (K=4, VALU): f1 = sin(30*(x@p1_w.T+p1_b)) -> S[:,0:32]
#pragma unroll
    for (int j = 0; j < 16; ++j) {
      const int c = 16 * hs + j;
      const float u = xv.x * Wf[FO_P1W + c * 4 + 0] + xv.y * Wf[FO_P1W + c * 4 + 1] +
                      xv.z * Wf[FO_P1W + c * 4 + 2] + xv.w * Wf[FO_P1W + c * 4 + 3] +
                      Wf[FO_P1B + c];
      S[l16 * SSTR + c] = (_Float16)__builtin_amdgcn_sinf(u * OMEGA_REV);
    }
    // ---- policy layer 2 on WMMA: f2 = sin(30*(f1@p2_w.T+p2_b)) -> S[:,32:64]
    gemm_stage<32, 32, P2_STR, SSTR, SSTR, 0>(S, Wh + OFF_WP2, Wf + FO_P2B, S + 32, nullptr, 0, lane);

    // ---- gate + experts as one K=288 WMMA chain ----
    // A = [enc_feat (H, k 0..255) | f2 (S cols 32..63, k 256..287)]
    {
      AV a[9];
#pragma unroll
      for (int ks = 0; ks < 8; ++ks) {
        const _Float16* ap = H + l16 * HSTR + 32 * ks + 8 * hs;
        a[ks].h[0] = *(const v8h*)(ap);
        a[ks].h[1] = *(const v8h*)(ap + 16);
      }
      {
        const _Float16* ap = S + l16 * SSTR + 32 + 8 * hs;
        a[8].h[0] = *(const v8h*)(ap);
        a[8].h[1] = *(const v8h*)(ap + 16);
      }
      v8f acc = {0.f, 0.f, 0.f, 0.f, 0.f, 0.f, 0.f, 0.f};
#pragma unroll
      for (int ks = 0; ks < 9; ++ks) {
        const _Float16* bp = Wh + OFF_WGE + (32 * ks + 8 * hs) * GE_STR + l16;
        AV b;
#pragma unroll
        for (int s = 0; s < 8; ++s) { b.v[s] = bp[s * GE_STR]; b.v[8 + s] = bp[(16 + s) * GE_STR]; }
        acc = __builtin_amdgcn_wmma_f32_16x16x32_f16(false, a[ks].v, false, b.v,
                                                     (short)0, acc, false, false);
      }
      // C: lane = logit index (0..6 gate, 7..13 experts), VGPR r = batch row.
      const float bv = Wf[FO_GEB + l16];
#pragma unroll
      for (int r = 0; r < 8; ++r)
        GE[(r + 8 * hs) * GSTR + l16] = acc[r] + bv;
    }

    // ---- per-row softmax(7) + soft routing (row = l16) ----
    float g[7], pr[7];
#pragma unroll
    for (int e = 0; e < 7; ++e) {
      g[e]  = GE[l16 * GSTR + e];
      pr[e] = GE[l16 * GSTR + 7 + e];
    }
    float m = g[0];
#pragma unroll
    for (int e = 1; e < 7; ++e) m = fmaxf(m, g[e]);
    float se = 0.f, y = 0.f;
#pragma unroll
    for (int e = 0; e < 7; ++e) {
      const float w = __expf(g[e] - m);
      se += w;
      y  += w * pr[e];
    }
    y /= se;
    if (hs == 0 && rowG < Btot) out[rowG] = y;
  }
}

extern "C" void kernel_launch(void* const* d_in, const int* in_sizes, int n_in,
                              void* d_out, int out_size, void* d_ws, size_t ws_size,
                              hipStream_t stream) {
  const float* x     = (const float*)d_in[0];
  const float* pe_w  = (const float*)d_in[1];
  const float* pe_b  = (const float*)d_in[2];
  const float* s1_w  = (const float*)d_in[3];
  const float* s1_b  = (const float*)d_in[4];
  const float* s2_w  = (const float*)d_in[5];
  const float* s2_b  = (const float*)d_in[6];
  const float* fc1_w = (const float*)d_in[7];
  const float* fc1_b = (const float*)d_in[8];
  const float* fc2_w = (const float*)d_in[9];
  const float* fc2_b = (const float*)d_in[10];
  const float* fc3_w = (const float*)d_in[11];
  const float* fc3_b = (const float*)d_in[12];
  const float* p1_w  = (const float*)d_in[13];
  const float* p1_b  = (const float*)d_in[14];
  const float* p2_w  = (const float*)d_in[15];
  const float* p2_b  = (const float*)d_in[16];
  const float* gate_w = (const float*)d_in[17];
  const float* gate_b = (const float*)d_in[18];
  const float* exp_w  = (const float*)d_in[19];
  const float* exp_b  = (const float*)d_in[20];

  const int Btot   = in_sizes[0] / 4;
  const int blocks = (Btot + ROWS_PER_BLOCK - 1) / ROWS_PER_BLOCK;

  (void)hipFuncSetAttribute((const void*)moe_inr_kernel,
                            hipFuncAttributeMaxDynamicSharedMemorySize,
                            (int)SMEM_BYTES);
  moe_inr_kernel<<<dim3(blocks), dim3(256), SMEM_BYTES, stream>>>(
      x, pe_w, pe_b, s1_w, s1_b, s2_w, s2_b, fc1_w, fc1_b, fc2_w, fc2_b,
      fc3_w, fc3_b, p1_w, p1_b, p2_w, p2_b, gate_w, gate_b, exp_w, exp_b,
      (float*)d_out, Btot);
}
